// StyleGAN2Generator_31190052503769
// MI455X (gfx1250) — matrix operand order repaired and checked by reading, not compiled, not run on hardware
//
#include <hip/hip_runtime.h>

// ---------------------------------------------------------------------------
// StyleGAN2 generator forward for gfx1250 (MI455X).
// Heavy 3x3 modulated convs -> implicit-GEMM with v_wmma_f32_16x16x32_f16.
// Modulation folded into input scaling, demodulation into output scaling.
// Weights pre-permuted to WMMA A-fragment order in global memory, so LDS
// staging is b128 global load -> b128 LDS store; fragment fetches are
// ds_load_b128; K-loop is double-buffered (stage k+1 while computing k).
// ---------------------------------------------------------------------------

typedef _Float16 h16;
typedef __attribute__((ext_vector_type(16))) _Float16 v16h;
typedef __attribute__((ext_vector_type(8)))  _Float16 v8h;
typedef __attribute__((ext_vector_type(8)))  float    v8f;

#define TB 256          // threads per block (8 waves of 32)
#define BM 128          // oc tile (GEMM M)
#define BN 64           // pixel tile (GEMM N)
#define BK 32           // K chunk (matches WMMA K)
#define WROW 40         // LDS row stride in halves (80B: conflict-free b128)

static __device__ __forceinline__ int clampi(int v, int lo, int hi) {
  return v < lo ? lo : (v > hi ? hi : v);
}
static __device__ __forceinline__ float leaky(float v) {
  return v > 0.f ? v : 0.2f * v;
}
// A-fragment slot permutation (ISA 7.12.2 16-bit A layout); involution.
static __device__ __forceinline__ int fragperm(int j) {
  const int m24 = j & 24;
  return (m24 == 8 || m24 == 16) ? (j ^ 24) : j;
}

// ---------------------------------------------------------------------------
// Style normalization: y = x * rsqrt(mean(x^2) + 1e-8), rows of 512.
// ---------------------------------------------------------------------------
__global__ __launch_bounds__(TB) void k_norm(const float* __restrict__ x,
                                             float* __restrict__ y) {
  __shared__ float red[TB];
  const int b = blockIdx.x;
  float s = 0.f;
  for (int i = threadIdx.x; i < 512; i += TB) {
    float v = x[b * 512 + i];
    s += v * v;
  }
  red[threadIdx.x] = s;
  __syncthreads();
  for (int st = TB / 2; st > 0; st >>= 1) {
    if (threadIdx.x < st) red[threadIdx.x] += red[threadIdx.x + st];
    __syncthreads();
  }
  const float r = rsqrtf(red[0] / 512.f + 1e-8f);
  for (int i = threadIdx.x; i < 512; i += TB) y[b * 512 + i] = x[b * 512 + i] * r;
}

// ---------------------------------------------------------------------------
// Small FC: y[b][o] = act(sum_i x[b][i] * W[o][i] + bias[o]).
// ---------------------------------------------------------------------------
__global__ __launch_bounds__(TB) void k_fc(const float* __restrict__ x,
                                           const float* __restrict__ W,
                                           const float* __restrict__ bias,
                                           float* __restrict__ y,
                                           int IN, int OUT, int do_leaky) {
  __shared__ float xs[512];
  const int b = blockIdx.y;
  for (int i = threadIdx.x; i < IN; i += TB) xs[i] = x[b * IN + i];
  __syncthreads();
  const int o = blockIdx.x * TB + threadIdx.x;
  if (o >= OUT) return;
  float acc = bias[o];
  const float* wr = W + (size_t)o * IN;
  for (int i = 0; i < IN; ++i) acc += xs[i] * wr[i];
  if (do_leaky) acc = leaky(acc);
  y[b * OUT + o] = acc;
}

// ---------------------------------------------------------------------------
// Weight prep: fp32 [oc][ic][3][3] -> f16 in chunked A-fragment order:
//   w16[[Kchunk][oc][32]] with slot = fragperm(K % 32), Kchunk = K/32, K=ic*9+k
// Also wsq[oc][ic] = sum_k w^2 for the demod reduction.
// ---------------------------------------------------------------------------
__global__ __launch_bounds__(TB) void k_wprep(const float* __restrict__ w,
                                              h16* __restrict__ w16,
                                              float* __restrict__ wsq,
                                              int OC, int IC) {
  const int i = blockIdx.x * TB + threadIdx.x;
  if (i >= OC * IC) return;
  const int oc = i / IC;
  const int ic = i - oc * IC;
  const float* wp = w + (size_t)i * 9;
  float sq = 0.f;
#pragma unroll
  for (int k = 0; k < 9; ++k) {
    const float v = wp[k];
    sq += v * v;
    const int Kg = ic * 9 + k;
    const int c = Kg >> 5;
    const int slot = fragperm(Kg & 31);
    w16[((size_t)c * OC + oc) * 32 + slot] = (h16)v;
  }
  wsq[i] = sq;
}

// demod[b][oc] = rsqrt(sum_ic wsq[oc][ic] * s[b][ic]^2 + 1e-8)
__global__ __launch_bounds__(TB) void k_demod(const float* __restrict__ wsq,
                                              const float* __restrict__ s,
                                              float* __restrict__ dem,
                                              int OC, int IC) {
  const int i = blockIdx.x * TB + threadIdx.x;
  if (i >= 8 * OC) return;
  const int b = i / OC;
  const int oc = i - b * OC;
  const float* wr = wsq + (size_t)oc * IC;
  const float* sr = s + b * IC;
  float acc = 0.f;
  for (int ic = 0; ic < IC; ++ic) {
    const float sv = sr[ic];
    acc += wr[ic] * sv * sv;
  }
  dem[i] = rsqrtf(acc + 1e-8f);
}

// ---------------------------------------------------------------------------
// Prepare const input: xmod[b][ic][4x4] = const[ic][p] * s[b][ic]  (f16)
// ---------------------------------------------------------------------------
__global__ __launch_bounds__(TB) void k_prep_const(const float* __restrict__ cst,
                                                   const float* __restrict__ s,
                                                   h16* __restrict__ xmod) {
  const int i = blockIdx.x * TB + threadIdx.x;
  if (i >= 8 * 512 * 16) return;
  const int p = i & 15;
  const int ic = (i >> 4) & 511;
  const int b = i >> 13;
  xmod[i] = (h16)(cst[ic * 16 + p] * s[b * 512 + ic]);
}

// ---------------------------------------------------------------------------
// Prepare layer input: optional 2x bilinear upsample (half-pixel, clamped,
// matching jax.image.resize) + per-(b,ic) modulation, f16 out.
// ---------------------------------------------------------------------------
__global__ __launch_bounds__(TB) void k_prepare(const h16* __restrict__ act,
                                                const float* __restrict__ s,
                                                h16* __restrict__ xmod,
                                                int IC, int Hin, int up) {
  const int Ho = up ? Hin * 2 : Hin;
  const size_t total = (size_t)8 * IC * Ho * Ho;
  const size_t i = (size_t)blockIdx.x * TB + threadIdx.x;
  if (i >= total) return;
  const int HW = Ho * Ho;
  const int p = (int)(i % HW);
  const int t = (int)(i / HW);
  const int ic = t % IC;
  const int b = t / IC;
  const h16* ab = act + ((size_t)b * IC + ic) * Hin * Hin;
  float v;
  if (up) {
    const int y = p / Ho, x = p - y * Ho;
    const float sy = (y + 0.5f) * 0.5f - 0.5f;
    const float sx = (x + 0.5f) * 0.5f - 0.5f;
    const int y0 = (int)floorf(sy);
    const int x0 = (int)floorf(sx);
    const float fy = sy - (float)y0;
    const float fx = sx - (float)x0;
    const int y0c = clampi(y0, 0, Hin - 1), y1c = clampi(y0 + 1, 0, Hin - 1);
    const int x0c = clampi(x0, 0, Hin - 1), x1c = clampi(x0 + 1, 0, Hin - 1);
    const float v00 = (float)ab[y0c * Hin + x0c];
    const float v01 = (float)ab[y0c * Hin + x1c];
    const float v10 = (float)ab[y1c * Hin + x0c];
    const float v11 = (float)ab[y1c * Hin + x1c];
    v = v00 * (1.f - fy) * (1.f - fx) + v01 * (1.f - fy) * fx +
        v10 * fy * (1.f - fx) + v11 * fy * fx;
  } else {
    v = (float)ab[p];
  }
  xmod[i] = (h16)(v * s[b * IC + ic]);
}

// ---------------------------------------------------------------------------
// Implicit-GEMM 3x3 conv with WMMA f32_16x16x32_f16, double-buffered LDS.
//   D[oc, pix] = sum_{k=ic*9} W[oc, k] * im2col(xmod)[k, pix]
// Block: 256 threads (8 waves). Tile 128(oc) x 64(pix), K chunks of 32.
// Weights arrive pre-permuted (chunked fragment order): each thread copies a
// contiguous 32B half-row per chunk (2x global_load_b128 + 2x ds_store_b128).
// The im2col gather fills the B rows (slot = fragperm(K), as for A). Fragment fetches
// are two 16B ds_load_b128 per operand (WROW=40 halves -> 20-word stride,
// conflict-free). Wave w: pixel sub-tile ni = w&3 (B reused x4), oc sub-tiles
// 4*(w>>2)..+3 -> 4 WMMAs / wave / K-chunk. Double-buffered: stage chunk k+1
// while computing k; one barrier per chunk; weight stream prefetched 2
// chunks ahead (global_prefetch_b8).
// Epilogue: *demod[b][oc] + noise_w*noise[pix] + bias[oc], leaky 0.2 -> f16.
// ---------------------------------------------------------------------------
__global__ __launch_bounds__(TB) void k_conv_wmma(
    const h16* __restrict__ xmod, const h16* __restrict__ w16,
    const float* __restrict__ demod, const float* __restrict__ noise,
    const float* __restrict__ bias, const float* __restrict__ noise_w,
    h16* __restrict__ out, int IC, int OC, int H) {
  const int HW = H * H;
  const int K = IC * 9;
  const int batch = blockIdx.z;
  const int ocbase = blockIdx.y * BM;
  const int pbase = blockIdx.x * BN;

  __shared__ __align__(16) h16 WsF[2][BM][WROW];  // A rows (fragment order)
  __shared__ __align__(16) h16 XsF[2][BN][WROW];  // B rows (fragment order)

  const int tid = threadIdx.x;
  const int lane = tid & 31;
  const int wave = tid >> 5;
  const int ni = wave & 3;             // pixel sub-tile of this wave
  const int miBase = (wave >> 2) * 4;  // first of 4 oc sub-tiles
  const int hl = lane >> 4;            // lane half (0/1)
  const int l15 = lane & 15;

  const h16* xb = xmod + (size_t)batch * IC * HW;
  // this thread's fixed slice of the weight copy: row wn, 32B half-row wh
  const int wn = tid >> 1;
  const int wh = tid & 1;

  // stage one K-chunk into LDS buffer `buf`
  auto stage = [&](int buf, int kb) {
    // --- weights: contiguous fragment-ordered rows; b128 copies, coalesced ---
    const h16* wrow = w16 + ((size_t)(kb >> 5) * OC + ocbase) * 32;
    const h16* gsrc = &wrow[(size_t)wn * 32 + wh * 16];
    if (kb + 2 * BK < K)
      __builtin_prefetch(&gsrc[(size_t)2 * OC * 32], 0, 1);
    const v8h w0 = *(const v8h*)&gsrc[0];
    const v8h w1 = *(const v8h*)&gsrc[8];
    *(v8h*)&WsF[buf][wn][wh * 16] = w0;
    *(v8h*)&WsF[buf][wn][wh * 16 + 8] = w1;
    // --- im2col gather of modulated input (B-fragment order, slot = fragperm(K)) ---
#pragma unroll
    for (int r = 0; r < (BK * BN) / TB; ++r) {
      const int idx = tid + r * TB;
      const int m = idx & (BN - 1);
      const int kk = idx >> 6;  // 0..31 == K within chunk
      const int k = kb + kk;
      const int ic = k / 9;
      const int t9 = k - ic * 9;
      const int dy = t9 / 3 - 1;
      const int dx = t9 - (t9 / 3) * 3 - 1;
      const int p = pbase + m;
      h16 v = (h16)0.f;
      if (p < HW) {
        const int y = p / H, x = p - y * H;
        const int sy = y + dy, sx = x + dx;
        if (sy >= 0 && sy < H && sx >= 0 && sx < H)
          v = xb[(size_t)ic * HW + sy * H + sx];
      }
      XsF[buf][m][fragperm(kk)] = v;
    }
  };

  v8f acc[4];
#pragma unroll
  for (int q = 0; q < 4; ++q) acc[q] = (v8f){0.f, 0.f, 0.f, 0.f, 0.f, 0.f, 0.f, 0.f};

  stage(0, 0);
  __syncthreads();

  int buf = 0;
  for (int kb = 0; kb < K; kb += BK) {
    // stage next chunk into the other buffer while computing this one
    if (kb + BK < K) stage(buf ^ 1, kb + BK);

    const int brow = ni * 16 + l15;
    const v8h blo = *(const v8h*)&XsF[buf][brow][hl * 16];
    const v8h bhi = *(const v8h*)&XsF[buf][brow][hl * 16 + 8];
    const v16h bf = __builtin_shufflevector(blo, bhi, 0, 1, 2, 3, 4, 5, 6, 7, 8,
                                            9, 10, 11, 12, 13, 14, 15);
#pragma unroll
    for (int q = 0; q < 4; ++q) {
      const int arow = (miBase + q) * 16 + l15;
      const v8h alo = *(const v8h*)&WsF[buf][arow][hl * 16];
      const v8h ahi = *(const v8h*)&WsF[buf][arow][hl * 16 + 8];
      const v16h af = __builtin_shufflevector(alo, ahi, 0, 1, 2, 3, 4, 5, 6, 7,
                                              8, 9, 10, 11, 12, 13, 14, 15);
      acc[q] = __builtin_amdgcn_wmma_f32_16x16x32_f16(false, af, false, bf,
                                                      (short)0, acc[q], false,
                                                      false);
    }
    __syncthreads();  // staging of buf^1 complete; reads of buf complete
    buf ^= 1;
  }

  // --- epilogue: demod + noise + bias + leaky, coalesced f16 stores ---
  const float nw = noise_w[0];
  const int p = pbase + ni * 16 + l15;  // D: N = lane&15
  if (p < HW) {
    const float nv = nw * noise[p];
#pragma unroll
    for (int q = 0; q < 4; ++q) {
#pragma unroll
      for (int r = 0; r < 8; ++r) {
        const int oc = ocbase + (miBase + q) * 16 + r + hl * 8;  // M=r+8*half
        const float v =
            leaky(acc[q][r] * demod[batch * OC + oc] + nv + bias[oc]);
        out[((size_t)batch * OC + oc) * HW + p] = (h16)v;
      }
    }
  }
}

// ---------------------------------------------------------------------------
// toRGB: 1x1 modulated conv (no demod) to 3 channels + bias + upsampled skip.
// ---------------------------------------------------------------------------
__global__ __launch_bounds__(TB) void k_torgb(const h16* __restrict__ act,
                                              const float* __restrict__ s,
                                              const float* __restrict__ w,
                                              const float* __restrict__ bias,
                                              const float* __restrict__ skip,
                                              float* __restrict__ outp,
                                              int IC, int H, int has_skip) {
  const int HW = H * H;
  const int i = blockIdx.x * TB + threadIdx.x;
  if (i >= 8 * HW) return;
  const int b = i / HW;
  const int p = i - b * HW;
  const h16* ab = act + (size_t)b * IC * HW + p;
  const float* sb = s + b * IC;
  float a0 = 0.f, a1 = 0.f, a2 = 0.f;
  for (int ic = 0; ic < IC; ++ic) {
    const float xs = (float)ab[(size_t)ic * HW] * sb[ic];
    a0 += w[ic] * xs;
    a1 += w[IC + ic] * xs;
    a2 += w[2 * IC + ic] * xs;
  }
  a0 += bias[0];
  a1 += bias[1];
  a2 += bias[2];
  if (has_skip) {
    const int Hs = H >> 1;
    const int y = p / H, x = p - y * H;
    const float sy = (y + 0.5f) * 0.5f - 0.5f;
    const float sx = (x + 0.5f) * 0.5f - 0.5f;
    const int y0 = (int)floorf(sy);
    const int x0 = (int)floorf(sx);
    const float fy = sy - (float)y0;
    const float fx = sx - (float)x0;
    const int y0c = clampi(y0, 0, Hs - 1), y1c = clampi(y0 + 1, 0, Hs - 1);
    const int x0c = clampi(x0, 0, Hs - 1), x1c = clampi(x0 + 1, 0, Hs - 1);
    float* accs[3] = {&a0, &a1, &a2};
#pragma unroll
    for (int c = 0; c < 3; ++c) {
      const float* sp = skip + ((size_t)b * 3 + c) * Hs * Hs;
      const float v00 = sp[y0c * Hs + x0c];
      const float v01 = sp[y0c * Hs + x1c];
      const float v10 = sp[y1c * Hs + x0c];
      const float v11 = sp[y1c * Hs + x1c];
      *accs[c] += v00 * (1.f - fy) * (1.f - fx) + v01 * (1.f - fy) * fx +
                  v10 * fy * (1.f - fx) + v11 * fy * fx;
    }
  }
  outp[((size_t)b * 3 + 0) * HW + p] = a0;
  outp[((size_t)b * 3 + 1) * HW + p] = a1;
  outp[((size_t)b * 3 + 2) * HW + p] = a2;
}

// ---------------------------------------------------------------------------
// Host orchestration.
// Input flattening assumed in setup_inputs() dict-insertion order:
//   0: styles
//   1..8:   mlp_w[0..7]        9..16: mlp_b[0..7]
//   17: const
//   18..22: conv1 {w, mod_w, mod_b, noise_w, bias}
//   23..26: rgb1  {w, mod_w, mod_b, bias}
//   27+5j:  convs[j] {w, mod_w, mod_b, noise_w, bias}, j=0..11
//   87+4j:  rgbs[j]  {w, mod_w, mod_b, bias},           j=0..5
//   111+l:  noises[l], l=0..12
// ---------------------------------------------------------------------------
extern "C" void kernel_launch(void* const* d_in, const int* in_sizes, int n_in,
                              void* d_out, int out_size, void* d_ws,
                              size_t ws_size, hipStream_t stream) {
  (void)in_sizes; (void)n_in; (void)out_size; (void)ws_size;

  const float* styles = (const float*)d_in[0];
  const float* mlp_w[8];
  const float* mlp_b[8];
  for (int i = 0; i < 8; ++i) {
    mlp_w[i] = (const float*)d_in[1 + i];
    mlp_b[i] = (const float*)d_in[9 + i];
  }
  const float* cst = (const float*)d_in[17];

  struct CP { const float *w, *mod_w, *mod_b, *noise_w, *bias; };
  struct RP { const float *w, *mod_w, *mod_b, *bias; };
  CP conv[13];
  RP rgb[7];
  conv[0] = {(const float*)d_in[18], (const float*)d_in[19],
             (const float*)d_in[20], (const float*)d_in[21],
             (const float*)d_in[22]};
  rgb[0] = {(const float*)d_in[23], (const float*)d_in[24],
            (const float*)d_in[25], (const float*)d_in[26]};
  for (int j = 0; j < 12; ++j) {
    const int base = 27 + 5 * j;
    conv[1 + j] = {(const float*)d_in[base + 0], (const float*)d_in[base + 1],
                   (const float*)d_in[base + 2], (const float*)d_in[base + 3],
                   (const float*)d_in[base + 4]};
  }
  for (int j = 0; j < 6; ++j) {
    const int base = 87 + 4 * j;
    rgb[1 + j] = {(const float*)d_in[base + 0], (const float*)d_in[base + 1],
                  (const float*)d_in[base + 2], (const float*)d_in[base + 3]};
  }
  const float* noise[13];
  for (int l = 0; l < 13; ++l) noise[l] = (const float*)d_in[111 + l];

  // --- carve workspace ---
  size_t off = 0;
  char* wsb = (char*)d_ws;
  auto carve = [&](size_t bytes) -> void* {
    void* p = wsb + off;
    off += (bytes + 255) & ~(size_t)255;
    return p;
  };
  float* latA = (float*)carve(8 * 512 * 4);
  float* latB = (float*)carve(8 * 512 * 4);
  float* Sbuf = (float*)carve(8 * 512 * 4);
  float* Dem  = (float*)carve(8 * 512 * 4);
  float* Wsq  = (float*)carve(512 * 512 * 4);
  h16*   W16  = (h16*)carve((size_t)4608 * 512 * 2);
  h16*   Xmod = (h16*)carve((size_t)8 * 256 * 65536 * 2);   // 268 MB max
  h16*   Act0 = (h16*)carve((size_t)8 * 128 * 65536 * 2);   // 134 MB max
  h16*   Act1 = (h16*)carve((size_t)8 * 128 * 65536 * 2);
  float* Skip0 = (float*)carve((size_t)8 * 3 * 16384 * 4);
  float* Skip1 = (float*)carve((size_t)8 * 3 * 16384 * 4);

  // --- mapping MLP ---
  k_norm<<<8, TB, 0, stream>>>(styles, latA);
  float* cur = latA;
  float* nxt = latB;
  for (int i = 0; i < 8; ++i) {
    k_fc<<<dim3(2, 8), TB, 0, stream>>>(cur, mlp_w[i], mlp_b[i], nxt, 512, 512, 1);
    float* t = cur; cur = nxt; nxt = t;
  }
  float* LAT = cur;  // [8,512] latent, identical across all latent slots

  // --- synthesis ---
  static const int convIC[13] = {512,512,512,512,512,512,512,512,512,512,256,256,128};
  static const int convOC[13] = {512,512,512,512,512,512,512,512,512,256,256,128,128};
  static const int convHin[13]= {4,  4,  8,  8, 16, 16, 32, 32, 64, 64,128,128,256};
  static const int convUp[13] = {0,  1,  0,  1,  0,  1,  0,  1,  0,  1,  0,  1,  0};

  h16* actCur = Act0;
  h16* actNxt = Act1;
  float* skipPrev = nullptr;
  int rgbIdx = 0;

  for (int l = 0; l < 13; ++l) {
    const int IC = convIC[l], OC = convOC[l];
    const int Hin = convHin[l], up = convUp[l];
    const int Ho = up ? 2 * Hin : Hin;
    const int HW = Ho * Ho;

    // style projection s[b][ic], weight prep, demod
    k_fc<<<dim3((IC + TB - 1) / TB, 8), TB, 0, stream>>>(
        LAT, conv[l].mod_w, conv[l].mod_b, Sbuf, 512, IC, 0);
    k_wprep<<<(OC * IC + TB - 1) / TB, TB, 0, stream>>>(conv[l].w, W16, Wsq, OC, IC);
    k_demod<<<(8 * OC + TB - 1) / TB, TB, 0, stream>>>(Wsq, Sbuf, Dem, OC, IC);

    // modulated (and optionally upsampled) f16 input
    if (l == 0) {
      k_prep_const<<<(8 * 512 * 16 + TB - 1) / TB, TB, 0, stream>>>(cst, Sbuf, Xmod);
    } else {
      const size_t tot = (size_t)8 * IC * HW;
      k_prepare<<<(unsigned)((tot + TB - 1) / TB), TB, 0, stream>>>(
          actCur, Sbuf, Xmod, IC, Hin, up);
    }

    // WMMA implicit-GEMM conv with fused epilogue
    k_conv_wmma<<<dim3((HW + BN - 1) / BN, OC / BM, 8), TB, 0, stream>>>(
        Xmod, W16, Dem, noise[l], conv[l].bias, conv[l].noise_w, actNxt, IC, OC, Ho);
    { h16* t = actCur; actCur = actNxt; actNxt = t; }

    // toRGB after conv1 and after every second conv
    if (l == 0 || (l >= 2 && (l % 2) == 0)) {
      const RP& rp = rgb[rgbIdx];
      k_fc<<<dim3((OC + TB - 1) / TB, 8), TB, 0, stream>>>(
          LAT, rp.mod_w, rp.mod_b, Sbuf, 512, OC, 0);
      float* dst = (rgbIdx == 6) ? (float*)d_out
                                 : ((rgbIdx & 1) ? Skip1 : Skip0);
      k_torgb<<<(8 * HW + TB - 1) / TB, TB, 0, stream>>>(
          actCur, Sbuf, rp.w, rp.bias, skipPrev, dst, OC, Ho, rgbIdx > 0);
      skipPrev = dst;
      ++rgbIdx;
    }
  }
}
